// LGBlock_37623913513239
// MI455X (gfx1250) — matrix operand order repaired and checked by reading, not compiled, not run on hardware
//
#include <hip/hip_runtime.h>
#include <hip/hip_bf16.h>
#include <math.h>

// ---------------------------------------------------------------------------
// Problem constants (from the reference)
// ---------------------------------------------------------------------------
#define D_   768
#define H_   12
#define HD_  64
#define B_   8
#define N_   128
#define S_   49
#define BN_  (B_ * N_)        // 1024
#define T_   (BN_ * S_)       // 50176 tokens total
#define TL_  (BN_ * (S_ - 1)) // 49152 "loc" tokens
#define TM_  BN_              // 1024 "msg" tokens
#define MLP_ (4 * D_)         // 3072

typedef __bf16 v8bf  __attribute__((ext_vector_type(8)));
typedef __bf16 v16bf __attribute__((ext_vector_type(16)));
typedef float  v8f   __attribute__((ext_vector_type(8)));

#define SHUF16(lo, hi) __builtin_shufflevector(lo, hi, 0,1,2,3,4,5,6,7,8,9,10,11,12,13,14,15)
#define WMMA_BF16(a, b, c) __builtin_amdgcn_wmma_f32_16x16x32_bf16(false, a, false, b, (short)0, c, false, false)

// A-fragment (16x32 bf16) load: lane holds row (l&15); K halves by (l>>4):
// elements = base + [0..15] (contiguous)
__device__ __forceinline__ v16bf load_afrag(const __bf16* p) {
    v8bf lo = *(const v8bf*)(p);
    v8bf hi = *(const v8bf*)(p + 8);
    return SHUF16(lo, hi);
}

// 16-lane-group reductions (xor masks 8,4,2,1 stay inside each half of wave32)
__device__ __forceinline__ float rowmax16(float x) {
#pragma unroll
    for (int off = 8; off; off >>= 1) x = fmaxf(x, __shfl_xor(x, off, 32));
    return x;
}
__device__ __forceinline__ float rowsum16(float x) {
#pragma unroll
    for (int off = 8; off; off >>= 1) x += __shfl_xor(x, off, 32);
    return x;
}

// ---------------------------------------------------------------------------
// GEMM: out[M,N] = A[M,K](bf16) @ W[N,K]^T(bf16) [+bias] [+res]  (fp32 accum)
// Block = 256 thr = 8 waves; wave tile = 32(M) x 64(N): 8 accumulators, each
// B fragment (one 32B contiguous load) feeds two WMMAs.
// Epilogue modes: 0 = f32 (+bias,+res)   1 = bf16 (+bias)   2 = gelu -> bf16
// ---------------------------------------------------------------------------
__global__ __launch_bounds__(256)
void gemm_bf16_wmma(const __bf16* __restrict__ A, const __bf16* __restrict__ W,
                    const float* __restrict__ bias, const float* __restrict__ res,
                    float* __restrict__ outf, __bf16* __restrict__ outb,
                    int M, int N, int K, int mode)
{
    const int lane  = threadIdx.x & 31;
    const int wave  = threadIdx.x >> 5;
    const int l15   = lane & 15;
    const int half  = lane >> 4;
    const int waveM = blockIdx.x * 256 + wave * 32;
    const int tileN = blockIdx.y * 64;

    v8f acc[2][4] = {};

    const __bf16* ar0 = A + (size_t)(waveM + l15) * K + half * 16;
    const __bf16* ar1 = ar0 + (size_t)16 * K;
    const __bf16* wr[4];
#pragma unroll
    for (int j = 0; j < 4; ++j)
        wr[j] = W + (size_t)(tileN + j * 16 + l15) * K + half * 16;

    for (int k0 = 0; k0 < K; k0 += 32) {
        v16bf a0 = load_afrag(ar0 + k0);
        v16bf a1 = load_afrag(ar1 + k0);
#pragma unroll
        for (int j = 0; j < 4; ++j) {
            v16bf b = *(const v16bf*)(wr[j] + k0);   // 32B contiguous
            acc[0][j] = WMMA_BF16(a0, b, acc[0][j]);
            acc[1][j] = WMMA_BF16(a1, b, acc[1][j]);
        }
    }

#pragma unroll
    for (int i = 0; i < 2; ++i) {
        const int rbase = waveM + i * 16 + half * 8;
#pragma unroll
        for (int j = 0; j < 4; ++j) {
            const int col = tileN + j * 16 + l15;
            const float bv = bias ? bias[col] : 0.0f;
#pragma unroll
            for (int r = 0; r < 8; ++r) {
                const size_t idx = (size_t)(rbase + r) * N + col;
                float v = acc[i][j][r] + bv;
                if (mode == 0) {
                    if (res) v += res[idx];
                    outf[idx] = v;
                } else if (mode == 1) {
                    outb[idx] = (__bf16)v;
                } else { // exact GELU
                    outb[idx] = (__bf16)(0.5f * v * (1.0f + erff(v * 0.70710678118654752f)));
                }
            }
        }
    }
}

// ---------------------------------------------------------------------------
// Flash attention on WMMA. One wave per (seq, head, 16-query tile).
// Q: bf16 [NS*T1, 768] (col h*64+d); KV: bf16 [NS*T2, 1536] (k@h*64, v@768+h*64)
// Vt: bf16 [(seq*H+h)*64 + d][T2p]  (pre-transposed, zero-padded to T2p)
// O: bf16 [NS*T1, 768].  SCALE = 0.125.
// Per 32-key chunk: 4 WMMA (QK^T) + online softmax on C-layout + P bounced
// through wave-private LDS (C->A layout) + 4 WMMA (P*V).
// ---------------------------------------------------------------------------
__global__ __launch_bounds__(256)
void flash_attn(const __bf16* __restrict__ Qb, const __bf16* __restrict__ KVb,
                const __bf16* __restrict__ Vt, __bf16* __restrict__ O,
                int T1, int T2, int T2p)
{
    __shared__ __align__(32) __bf16 pbuf[8][16 * 32]; // 1KB per wave
    const int lane = threadIdx.x & 31;
    const int wv   = threadIdx.x >> 5;
    const int l15  = lane & 15;
    const int half = lane >> 4;
    const int nqb  = (T1 + 15) >> 4;
    const int wid  = (int)(blockIdx.x * 8 + wv);
    const int qb   = wid % nqb;
    const int hh   = (wid / nqb) % H_;
    const int seq  = wid / (nqb * H_);

    // Q fragments for the two 32-wide d-chunks (clamp padded query rows)
    int t1r = qb * 16 + l15; if (t1r > T1 - 1) t1r = T1 - 1;
    const __bf16* qrow = Qb + (size_t)(seq * T1 + t1r) * D_ + hh * HD_ + half * 16;
    v16bf aq0 = load_afrag(qrow);
    v16bf aq1 = load_afrag(qrow + 32);

    v8f ot[4] = {};
    float mrow[8], lrow[8];
#pragma unroll
    for (int r = 0; r < 8; ++r) { mrow[r] = -1e30f; lrow[r] = 0.f; }

    const __bf16* kvb = KVb + (size_t)seq * T2 * (2 * D_) + hh * HD_;
    const __bf16* vtb = Vt + (size_t)(seq * H_ + hh) * HD_ * T2p;
    __bf16* pb = pbuf[wv];

    for (int kc = 0; kc < T2p; kc += 32) {
        // ---- scores: two 16-key tiles, contraction over d=64 ----
        v8f st[2];
#pragma unroll
        for (int j = 0; j < 2; ++j) {
            int krow = kc + j * 16 + l15; if (krow > T2 - 1) krow = T2 - 1;
            const __bf16* kr = kvb + (size_t)krow * (2 * D_) + half * 16;
            v16bf b0 = *(const v16bf*)(kr);
            v16bf b1 = *(const v16bf*)(kr + 32);
            v8f s = {};
            s = WMMA_BF16(aq0, b0, s);
            s = WMMA_BF16(aq1, b1, s);
            st[j] = s;
        }
        const bool msk0 = (kc + l15)      >= T2;   // lane = key column
        const bool msk1 = (kc + 16 + l15) >= T2;
#pragma unroll
        for (int r = 0; r < 8; ++r) {
            st[0][r] = msk0 ? -1e30f : st[0][r] * 0.125f;
            st[1][r] = msk1 ? -1e30f : st[1][r] * 0.125f;
        }
        // ---- online softmax (row r of C-tile = query r + 8*half) ----
        float e0[8], e1[8], co[8];
#pragma unroll
        for (int r = 0; r < 8; ++r) {
            float cm = rowmax16(fmaxf(st[0][r], st[1][r]));
            float mn = fmaxf(mrow[r], cm);
            co[r] = __expf(mrow[r] - mn);
            e0[r] = __expf(st[0][r] - mn);
            e1[r] = __expf(st[1][r] - mn);
            float cs = rowsum16(e0[r] + e1[r]);
            lrow[r] = lrow[r] * co[r] + cs;
            mrow[r] = mn;
        }
#pragma unroll
        for (int t = 0; t < 4; ++t)
#pragma unroll
            for (int r = 0; r < 8; ++r) ot[t][r] *= co[r];

        // ---- P: C-layout -> A-layout via wave-private LDS bounce ----
#pragma unroll
        for (int r = 0; r < 8; ++r) {
            const int row = r + 8 * half;
            pb[row * 32 + l15]      = (__bf16)e0[r];
            pb[row * 32 + 16 + l15] = (__bf16)e1[r];
        }
        // per-wave LDS ops are in-order; fence keeps the compiler honest too
        asm volatile("s_wait_dscnt 0" ::: "memory");
        v16bf ap = load_afrag(pb + l15 * 32 + half * 16);

        // ---- O += P @ V : 4 d-tiles, contraction over 32 keys ----
#pragma unroll
        for (int t = 0; t < 4; ++t) {
            const __bf16* vr = vtb + (size_t)(t * 16 + l15) * T2p + kc + half * 16;
            v16bf bv = *(const v16bf*)(vr);
            ot[t] = WMMA_BF16(ap, bv, ot[t]);
        }
    }

    // ---- normalize + store (mask padded query rows) ----
#pragma unroll
    for (int r = 0; r < 8; ++r) {
        const int q = qb * 16 + r + 8 * half;
        if (q < T1) {
            const float inv = 1.0f / lrow[r];
            __bf16* o = O + (size_t)(seq * T1 + q) * D_ + hh * HD_;
#pragma unroll
            for (int t = 0; t < 4; ++t)
                o[t * 16 + l15] = (__bf16)(ot[t][r] * inv);
        }
    }
}

// ---------------------------------------------------------------------------
// V transpose: Vt[(seq*H+h)*64 + d][t2p] = t2p<T2 ? KV[seq*T2+t2p][768+h*64+d] : 0
// ---------------------------------------------------------------------------
__global__ void transpose_v(const __bf16* __restrict__ KV, __bf16* __restrict__ Vt,
                            int T2, int T2p)
{
    const size_t idx = (size_t)blockIdx.x * 256 + threadIdx.x;
    const size_t row = idx / T2p;
    const int    t2  = (int)(idx % T2p);
    const size_t d   = row % HD_;
    const size_t sh  = row / HD_;          // seq*H + h
    const size_t h   = sh % H_;
    const size_t seq = sh / H_;
    __bf16 v = (__bf16)0.f;
    if (t2 < T2)
        v = KV[(seq * T2 + t2) * (size_t)(2 * D_) + D_ + h * HD_ + d];
    Vt[idx] = v;
}

// ---------------------------------------------------------------------------
// LayerNorm over D=768 -> bf16 output. One block per row.
// ---------------------------------------------------------------------------
__global__ __launch_bounds__(256)
void ln_to_bf16(const float* __restrict__ X, const float* __restrict__ g,
                const float* __restrict__ bb, __bf16* __restrict__ Y)
{
    const int row = blockIdx.x;
    const float* x = X + (size_t)row * D_;
    float v[3], s = 0.f, s2 = 0.f;
#pragma unroll
    for (int i = 0; i < 3; ++i) {
        v[i] = x[threadIdx.x + i * 256];
        s += v[i]; s2 += v[i] * v[i];
    }
#pragma unroll
    for (int off = 16; off; off >>= 1) {
        s  += __shfl_xor(s,  off, 32);
        s2 += __shfl_xor(s2, off, 32);
    }
    __shared__ float ws[8], ws2[8];
    const int wv = threadIdx.x >> 5, ln = threadIdx.x & 31;
    if (ln == 0) { ws[wv] = s; ws2[wv] = s2; }
    __syncthreads();
    s = 0.f; s2 = 0.f;
#pragma unroll
    for (int w = 0; w < 8; ++w) { s += ws[w]; s2 += ws2[w]; }
    const float mean = s * (1.0f / D_);
    const float var  = s2 * (1.0f / D_) - mean * mean;
    const float inv  = rsqrtf(var + 1e-5f);
#pragma unroll
    for (int i = 0; i < 3; ++i) {
        const int c = threadIdx.x + i * 256;
        Y[(size_t)row * D_ + c] = (__bf16)((v[i] - mean) * inv * g[c] + bb[c]);
    }
}

// ---------------------------------------------------------------------------
// Elementwise helpers
// ---------------------------------------------------------------------------
__global__ void cvt_f32_to_bf16(const float* __restrict__ in, __bf16* __restrict__ out)
{
    const size_t i = (size_t)blockIdx.x * blockDim.x + threadIdx.x;
    out[i] = (__bf16)in[i];
}

__global__ void gather_msg(const float* __restrict__ X, float* __restrict__ M)
{
    const size_t idx = (size_t)blockIdx.x * blockDim.x + threadIdx.x;
    const size_t r = idx / D_, c = idx % D_;
    M[idx] = X[(r * S_) * D_ + c];
}
__global__ void scatter_msg(const float* __restrict__ M, float* __restrict__ X)
{
    const size_t idx = (size_t)blockIdx.x * blockDim.x + threadIdx.x;
    const size_t r = idx / D_, c = idx % D_;
    X[(r * S_) * D_ + c] = M[idx];
}
__global__ void gather_loc(const float* __restrict__ X, float* __restrict__ L)
{
    const size_t idx = (size_t)blockIdx.x * blockDim.x + threadIdx.x;
    const size_t r = idx / D_, c = idx % D_;
    const size_t b = r / (size_t)(N_ * (S_ - 1));
    const size_t j = r % (size_t)(N_ * (S_ - 1));
    const size_t n = j / (S_ - 1), s = j % (S_ - 1) + 1;
    L[idx] = X[((b * N_ + n) * S_ + s) * D_ + c];
}
__global__ void scatter_loc(const float* __restrict__ L, float* __restrict__ X)
{
    const size_t idx = (size_t)blockIdx.x * blockDim.x + threadIdx.x;
    const size_t r = idx / D_, c = idx % D_;
    const size_t b = r / (size_t)(N_ * (S_ - 1));
    const size_t j = r % (size_t)(N_ * (S_ - 1));
    const size_t n = j / (S_ - 1), s = j % (S_ - 1) + 1;
    X[((b * N_ + n) * S_ + s) * D_ + c] = L[idx];
}

// ---------------------------------------------------------------------------
// Host orchestration
// ---------------------------------------------------------------------------
extern "C" void kernel_launch(void* const* d_in, const int* in_sizes, int n_in,
                              void* d_out, int out_size, void* d_ws, size_t ws_size,
                              hipStream_t stream)
{
    (void)in_sizes; (void)n_in; (void)out_size; (void)ws_size;

    const float* x = (const float*)d_in[0];
    const float* aw[4][4]; // [attn][qw,kvw,pw,pb]
    for (int i = 0; i < 4; ++i)
        for (int j = 0; j < 4; ++j)
            aw[i][j] = (const float*)d_in[2 + i * 4 + j];
    const float* ln_f0_g = (const float*)d_in[18]; const float* ln_f0_b = (const float*)d_in[19];
    const float* ln_s0_g = (const float*)d_in[20]; const float* ln_s0_b = (const float*)d_in[21];
    const float* ln_t0_g = (const float*)d_in[22]; const float* ln_t0_b = (const float*)d_in[23];
    const float* ln_t1_g = (const float*)d_in[24]; const float* ln_t1_b = (const float*)d_in[25];
    const float* ln_q0_g = (const float*)d_in[26]; const float* ln_q0_b = (const float*)d_in[27];
    const float* ln_q1_g = (const float*)d_in[28]; const float* ln_q1_b = (const float*)d_in[29];
    const float* ln2_g   = (const float*)d_in[30]; const float* ln2_b   = (const float*)d_in[31];
    const float* fc1_w = (const float*)d_in[32]; const float* fc1_b = (const float*)d_in[33];
    const float* fc2_w = (const float*)d_in[34]; const float* fc2_b = (const float*)d_in[35];

    float* xo = (float*)d_out; // running residual stream x [T_, 768] f32

    // ---- workspace bump allocator -------------------------------------
    char* p = (char*)d_ws;
    auto alloc = [&](size_t bytes) -> char* {
        char* r = p; p += (bytes + 255) & ~(size_t)255; return r;
    };
    __bf16* Qb  = (__bf16*)alloc((size_t)T_ * D_ * 2);            // q (bf16)
    __bf16* KVb = (__bf16*)alloc((size_t)T_ * 2 * D_ * 2);        // kv (bf16)
    __bf16* Vtb = (__bf16*)alloc((size_t)BN_ * H_ * HD_ * 64 * 2);// V^T (max: stage A)
    __bf16* Hb  = (__bf16*)alloc((size_t)T_ * MLP_ * 2);          // LN/attn-O/gelu staging
    __bf16* Gb  = (__bf16*)alloc((size_t)T_ * D_ * 2);            // LN2 output
    float*  Lf  = (float*)alloc((size_t)TL_ * D_ * 4);
    float*  Mf  = (float*)alloc((size_t)TM_ * D_ * 4);
    __bf16* Mb  = (__bf16*)alloc((size_t)TM_ * D_ * 2);
    __bf16* Wb  = (__bf16*)alloc((size_t)(4 * (4 * D_ * D_) + 2 * MLP_ * D_) * 2);

    // ---- convert weights to bf16 (28 MB -> L2-resident) ----------------
    __bf16* wq[4]; __bf16* wkv[4]; __bf16* wp[4]; __bf16* wfc1; __bf16* wfc2;
    {
        __bf16* q = Wb;
        auto cvt = [&](const float* src, size_t n) -> __bf16* {
            __bf16* dst = q; q += n;
            cvt_f32_to_bf16<<<dim3((unsigned)(n / 256)), 256, 0, stream>>>(src, dst);
            return dst;
        };
        for (int i = 0; i < 4; ++i) {
            wq[i]  = cvt(aw[i][0], (size_t)D_ * D_);
            wkv[i] = cvt(aw[i][1], (size_t)2 * D_ * D_);
            wp[i]  = cvt(aw[i][2], (size_t)D_ * D_);
        }
        wfc1 = cvt(fc1_w, (size_t)MLP_ * D_);
        wfc2 = cvt(fc2_w, (size_t)D_ * MLP_);
    }

    auto gemmF = [&](const __bf16* A, const __bf16* W, const float* bias,
                     const float* res, float* out, int M, int N, int K) {
        gemm_bf16_wmma<<<dim3((unsigned)(M / 256), (unsigned)(N / 64)), 256, 0, stream>>>(
            A, W, bias, res, out, nullptr, M, N, K, 0);
    };
    auto gemmB = [&](const __bf16* A, const __bf16* W, __bf16* out, int M, int N, int K) {
        gemm_bf16_wmma<<<dim3((unsigned)(M / 256), (unsigned)(N / 64)), 256, 0, stream>>>(
            A, W, nullptr, nullptr, nullptr, out, M, N, K, 1);
    };
    auto gemmG = [&](const __bf16* A, const __bf16* W, const float* bias,
                     __bf16* out, int M, int N, int K) {
        gemm_bf16_wmma<<<dim3((unsigned)(M / 256), (unsigned)(N / 64)), 256, 0, stream>>>(
            A, W, bias, nullptr, nullptr, out, M, N, K, 2);
    };
    auto vtrans = [&](int NS, int T2, int T2p) {
        transpose_v<<<dim3((unsigned)((size_t)NS * H_ * HD_ * T2p / 256)), 256, 0, stream>>>(
            KVb, Vtb, T2, T2p);
    };
    auto flash = [&](const __bf16* Q, __bf16* O, int NS, int T1, int T2, int T2p) {
        const unsigned waves = (unsigned)((size_t)NS * H_ * ((T1 + 15) / 16));
        flash_attn<<<dim3(waves / 8), 256, 0, stream>>>(Q, KVb, Vtb, O, T1, T2, T2p);
    };

    // ============ stage A: attn1 over all tokens (1024 seqs, T=49) ======
    hipMemcpyAsync(xo, x, (size_t)T_ * D_ * 4, hipMemcpyDeviceToDevice, stream);
    ln_to_bf16<<<T_, 256, 0, stream>>>(xo, ln_f0_g, ln_f0_b, Hb);
    gemmB(Hb, wq[0],  Qb,  T_, D_,     D_);
    gemmB(Hb, wkv[0], KVb, T_, 2 * D_, D_);
    vtrans(BN_, S_, 64);
    flash(Qb, Hb, BN_, S_, S_, 64);
    gemmF(Hb, wp[0], aw[0][3], xo, xo, T_, D_, D_);      // x += proj

    gather_msg<<<(TM_ * D_) / 256, 256, 0, stream>>>(xo, Mf);
    gather_loc<<<(TL_ * D_) / 256, 256, 0, stream>>>(xo, Lf);

    // ============ stage B: attn2 over msg (8 seqs, T=128) ===============
    ln_to_bf16<<<TM_, 256, 0, stream>>>(Mf, ln_s0_g, ln_s0_b, Mb);
    gemmB(Mb, wq[1],  Qb,  TM_, D_,     D_);
    gemmB(Mb, wkv[1], KVb, TM_, 2 * D_, D_);
    vtrans(B_, N_, N_);
    flash(Qb, Mb, B_, N_, N_, N_);
    gemmF(Mb, wp[1], aw[1][3], Mf, Mf, TM_, D_, D_);     // msg += proj

    // ============ stage C: attn3 loc<-msg (q:6144, kv:128) ==============
    ln_to_bf16<<<TL_, 256, 0, stream>>>(Lf, ln_t0_g, ln_t0_b, Hb);
    gemmB(Hb, wq[2], Qb, TL_, D_, D_);
    ln_to_bf16<<<TM_, 256, 0, stream>>>(Mf, ln_t1_g, ln_t1_b, Mb);
    gemmB(Mb, wkv[2], KVb, TM_, 2 * D_, D_);
    vtrans(B_, N_, N_);
    flash(Qb, Hb, B_, N_ * (S_ - 1), N_, N_);
    gemmF(Hb, wp[2], aw[2][3], Lf, Lf, TL_, D_, D_);     // loc += proj

    // ============ stage D: attn4 msg<-loc (q:128, kv:6144) ==============
    ln_to_bf16<<<TM_, 256, 0, stream>>>(Mf, ln_q0_g, ln_q0_b, Mb);
    gemmB(Mb, wq[3], Qb, TM_, D_, D_);
    ln_to_bf16<<<TL_, 256, 0, stream>>>(Lf, ln_q1_g, ln_q1_b, Hb);
    gemmB(Hb, wkv[3], KVb, TL_, 2 * D_, D_);
    vtrans(B_, N_ * (S_ - 1), N_ * (S_ - 1));
    flash(Qb, Mb, B_, N_, N_ * (S_ - 1), N_ * (S_ - 1));
    gemmF(Mb, wp[3], aw[3][3], Mf, Mf, TM_, D_, D_);     // msg += proj

    scatter_msg<<<(TM_ * D_) / 256, 256, 0, stream>>>(Mf, xo);
    scatter_loc<<<(TL_ * D_) / 256, 256, 0, stream>>>(Lf, xo);

    // ============ stage E: MLP, GELU fused into fc1 epilogue ============
    ln_to_bf16<<<T_, 256, 0, stream>>>(xo, ln2_g, ln2_b, Gb);
    gemmG(Gb, wfc1, fc1_b, Hb, T_, MLP_, D_);            // Hb = gelu(fc1) bf16
    gemmF(Hb, wfc2, fc2_b, xo, xo, T_, D_, MLP_);        // x += fc2
}
